// DistSparseMoe_21775484191499
// MI455X (gfx1250) — compile-verified, matrix operand order checked by reading, not run on hardware
//
#include <hip/hip_runtime.h>

// ---------------- Types for CDNA5 WMMA ----------------
typedef __bf16 bf16_t;
typedef bf16_t v16bf __attribute__((ext_vector_type(16)));
typedef bf16_t v2bf  __attribute__((ext_vector_type(2)));
typedef float  v8f   __attribute__((ext_vector_type(8)));

union Frag { v16bf v; unsigned int u[8]; };
union F4   { float4 v; float f[4]; };

// pack two floats into one dword of bf16 {lo, hi} using native bf16 cvt (RTE)
static __device__ __forceinline__ unsigned int pack2bf(float lo, float hi) {
    union { v2bf v; unsigned int u; } c;
    c.v = (v2bf){(bf16_t)lo, (bf16_t)hi};
    return c.u;
}

// Problem constants (match reference setup_inputs)
#define T_TOK 32768   // B*S
#define HDIM  1024
#define NEXP  8
#define NCHUNK 4096   // T_TOK / 8 tokens per chunk

// ================= Kernel 1: router + softmax + argmax + chunk hist ==========
__global__ __launch_bounds__(256) void moe_router(
    const float* __restrict__ x, const float* __restrict__ Wr,
    const float* __restrict__ br, int* __restrict__ best,
    float* __restrict__ bestp, int* __restrict__ chunkCnt) {
    __shared__ int ldsCnt[NEXP];
    const int tid = threadIdx.x;
    if (tid < NEXP) ldsCnt[tid] = 0;
    __syncthreads();

    const int wave = tid >> 5, lane = tid & 31;
    const int t = blockIdx.x * 8 + wave;

    float acc[NEXP];
#pragma unroll
    for (int e = 0; e < NEXP; ++e) acc[e] = 0.0f;

    const float* xr = x + (size_t)t * HDIM;
    for (int hh = lane; hh < HDIM; hh += 32) {
        const float xv = xr[hh];
        const float* wr = Wr + hh * NEXP;
#pragma unroll
        for (int e = 0; e < NEXP; ++e) acc[e] += xv * wr[e];
    }
#pragma unroll
    for (int off = 16; off >= 1; off >>= 1) {
#pragma unroll
        for (int e = 0; e < NEXP; ++e) acc[e] += __shfl_xor(acc[e], off, 32);
    }
    if (lane == 0) {
        float mx = -3.4e38f; int bi = 0;
#pragma unroll
        for (int e = 0; e < NEXP; ++e) {
            const float l = acc[e] + br[e];
            acc[e] = l;
            if (l > mx) { mx = l; bi = e; }   // first-max wins, matches argmax
        }
        float sum = 0.0f;
#pragma unroll
        for (int e = 0; e < NEXP; ++e) sum += __expf(acc[e] - mx);
        best[t] = bi;
        bestp[t] = 1.0f / sum;                // softmax prob of best expert
        atomicAdd(&ldsCnt[bi], 1);
    }
    __syncthreads();
    if (tid < NEXP) chunkCnt[tid * NCHUNK + blockIdx.x] = ldsCnt[tid];
}

// ================= Kernel 2: per-expert exclusive scan over 4096 chunks ======
__global__ __launch_bounds__(256) void moe_scan(
    const int* __restrict__ chunkCnt, int* __restrict__ chunkPre,
    int* __restrict__ totals) {
    const int e = blockIdx.x;
    const int* cnt = chunkCnt + e * NCHUNK;
    int* pre = chunkPre + e * NCHUNK;
    __shared__ int lds[256];
    const int tid = threadIdx.x;
    const int base = tid * 16;

    int local[16]; int s = 0;
#pragma unroll
    for (int i = 0; i < 16; ++i) { local[i] = s; s += cnt[base + i]; }
    lds[tid] = s;
    __syncthreads();
    for (int off = 1; off < 256; off <<= 1) {
        const int v = (tid >= off) ? lds[tid - off] : 0;
        __syncthreads();
        lds[tid] += v;
        __syncthreads();
    }
    const int incl = lds[tid];
    const int excl = incl - s;
#pragma unroll
    for (int i = 0; i < 16; ++i) pre[base + i] = excl + local[i];
    if (tid == 255) totals[e] = incl;
}

// ================= Kernel 3: expert base offsets (tiny) ======================
__global__ void moe_offsets(const int* __restrict__ totals, int* __restrict__ offs) {
    if (threadIdx.x == 0) {
        int s = 0;
        for (int e = 0; e < NEXP; ++e) { offs[e] = s; s += totals[e]; }
    }
}

// ================= Kernel 4: stable scatter -> src permutation ===============
__global__ void moe_scatter(const int* __restrict__ best,
                            const int* __restrict__ chunkPre,
                            const int* __restrict__ offs,
                            int* __restrict__ srcIdx) {
    if (threadIdx.x != 0) return;
    const int chunk = blockIdx.x;
    int cnt[NEXP];
#pragma unroll
    for (int e = 0; e < NEXP; ++e) cnt[e] = 0;
    for (int i = 0; i < 8; ++i) {
        const int t = chunk * 8 + i;
        const int e = best[t];
        const int p = offs[e] + chunkPre[e * NCHUNK + chunk] + cnt[e]++;
        srcIdx[p] = t;
    }
}

// ================= Kernel 5: gathered GEMM via bf16 WMMA =====================
// out[p, :] = (h[src[p], :] @ W + b) * bestp[p]
#define BM 128
#define BN 128
#define BK 32
#define LDT 40   // padded LDS row stride in bf16 (80B: 8B-aligned, conflict-free)

__global__ __launch_bounds__(256) void moe_gemm(
    const float* __restrict__ h, const float* __restrict__ W,
    const float* __restrict__ b, const int* __restrict__ srcIdx,
    const float* __restrict__ bestp, float* __restrict__ out) {
    __shared__ unsigned short sA[BM * LDT];   // A tile, [m][k] bf16
    __shared__ unsigned short sB[BN * LDT];   // B tile transposed, [n][k] bf16

    const int tid  = threadIdx.x;
    const int wave = tid >> 5;
    const int lane = tid & 31;
    const int l16  = lane & 15;
    const int hlf  = lane >> 4;

    const int p0 = blockIdx.x * BM;
    const int n0 = blockIdx.y * BN;
    const int wm = (wave & 1) * 64;    // wave row offset in tile
    const int wn = (wave >> 1) * 32;   // wave col offset in tile

    // ---- K-invariant staging descriptors ----
    // A: 1024 float4 units: unit u -> row = u>>3 (0..127), c4 = u&7 (0..7)
    const float* aSrc[4];
    unsigned short* aDst[4];
#pragma unroll
    for (int i = 0; i < 4; ++i) {
        const int u = tid + i * 256;
        const int row = u >> 3, c4 = u & 7;
        aSrc[i] = h + (size_t)srcIdx[p0 + row] * HDIM + c4 * 4;
        aDst[i] = &sA[row * LDT + c4 * 4];
    }
    // B: 512 k-pair units: unit u -> kp = u>>5 (0..15), c4 = u&31 (0..31)
    const float* bSrc[2];
    unsigned short* bDst[2];
#pragma unroll
    for (int i = 0; i < 2; ++i) {
        const int u = tid + i * 256;
        const int kp = u >> 5, c4 = u & 31;
        bSrc[i] = W + (size_t)(2 * kp) * HDIM + n0 + c4 * 4;
        bDst[i] = &sB[(c4 * 4) * LDT + 2 * kp];
    }

    // K offsets per fragment dword, per documented 16-bit A-layout
    int kd[8];
#pragma unroll
    for (int d = 0; d < 8; ++d) kd[d] = ((d & 4) ? 16 : 0) + hlf * 8 + ((d & 3) << 1);

    v8f acc[4][2];
#pragma unroll
    for (int mt = 0; mt < 4; ++mt)
#pragma unroll
        for (int nt = 0; nt < 2; ++nt)
            acc[mt][nt] = (v8f){0.f, 0.f, 0.f, 0.f, 0.f, 0.f, 0.f, 0.f};

    // ---- prologue: load first tile into staging registers ----
    F4 rA[4], rB[2][2];
#pragma unroll
    for (int i = 0; i < 4; ++i) rA[i].v = *(const float4*)(aSrc[i]);
#pragma unroll
    for (int i = 0; i < 2; ++i) {
        rB[i][0].v = *(const float4*)(bSrc[i]);
        rB[i][1].v = *(const float4*)(bSrc[i] + HDIM);
    }

    for (int kk = 0; kk < HDIM; kk += BK) {
        // ---- convert fp32 -> bf16 (native cvt), packed LDS stores ----
#pragma unroll
        for (int i = 0; i < 4; ++i) {       // A: one b64 store per float4
            uint2 d;
            d.x = pack2bf(rA[i].f[0], rA[i].f[1]);
            d.y = pack2bf(rA[i].f[2], rA[i].f[3]);
            *(uint2*)(aDst[i]) = d;
        }
#pragma unroll
        for (int i = 0; i < 2; ++i) {       // B: 4 b32 stores (k-pair per dword)
#pragma unroll
            for (int j = 0; j < 4; ++j) {
                *(unsigned int*)(bDst[i] + j * LDT) =
                    pack2bf(rB[i][0].f[j], rB[i][1].f[j]);
            }
        }
        __syncthreads();

        // ---- prefetch next tile into registers (overlaps with WMMAs) ----
        if (kk + BK < HDIM) {
            const int kn = kk + BK;
#pragma unroll
            for (int i = 0; i < 4; ++i) rA[i].v = *(const float4*)(aSrc[i] + kn);
#pragma unroll
            for (int i = 0; i < 2; ++i) {
                const float* p = bSrc[i] + (size_t)kn * HDIM;
                rB[i][0].v = *(const float4*)(p);
                rB[i][1].v = *(const float4*)(p + HDIM);
            }
        }

        // ---- fragments + 8 WMMAs ----
        Frag bfr[2];
#pragma unroll
        for (int nt = 0; nt < 2; ++nt) {
            const unsigned short* bp = &sB[(wn + nt * 16 + l16) * LDT];
#pragma unroll
            for (int d = 0; d < 8; ++d)
                bfr[nt].u[d] = *(const unsigned int*)(bp + kd[d]);
        }
#pragma unroll
        for (int mt = 0; mt < 4; ++mt) {
            const unsigned short* ap = &sA[(wm + mt * 16 + l16) * LDT];
            Frag af;
#pragma unroll
            for (int d = 0; d < 8; ++d)
                af.u[d] = *(const unsigned int*)(ap + kd[d]);
#pragma unroll
            for (int nt = 0; nt < 2; ++nt) {
                acc[mt][nt] = __builtin_amdgcn_wmma_f32_16x16x32_bf16(
                    false, af.v, false, bfr[nt].v,
                    (short)0, acc[mt][nt], false, false);
            }
        }
        __syncthreads();
    }

    // ---- epilogue: +bias, * bestp (original-order prob), store fp32 ----
    float bias[2];
#pragma unroll
    for (int nt = 0; nt < 2; ++nt) bias[nt] = b[n0 + wn + nt * 16 + l16];

#pragma unroll
    for (int mt = 0; mt < 4; ++mt) {
#pragma unroll
        for (int r = 0; r < 8; ++r) {
            const int p = p0 + wm + mt * 16 + hlf * 8 + r;
            const float sc = bestp[p];
#pragma unroll
            for (int nt = 0; nt < 2; ++nt) {
                const int j = n0 + wn + nt * 16 + l16;
                out[(size_t)p * HDIM + j] = (acc[mt][nt][r] + bias[nt]) * sc;
            }
        }
    }
}

// =============================== launcher ====================================
extern "C" void kernel_launch(void* const* d_in, const int* in_sizes, int n_in,
                              void* d_out, int out_size, void* d_ws, size_t ws_size,
                              hipStream_t stream) {
    const float* x  = (const float*)d_in[0];   // [T, H]
    const float* Wr = (const float*)d_in[1];   // [H, E]
    const float* br = (const float*)d_in[2];   // [E]
    const float* We = (const float*)d_in[3];   // [H, H]
    const float* be = (const float*)d_in[4];   // [H]
    float* out = (float*)d_out;

    char* ws = (char*)d_ws;
    int*   best     = (int*)(ws + 0);                 // 128K
    float* bestp    = (float*)(ws + (128 << 10));     // 128K
    int*   chunkCnt = (int*)(ws + (256 << 10));       // 128K
    int*   chunkPre = (int*)(ws + (384 << 10));       // 128K
    int*   totals   = (int*)(ws + (512 << 10));       // 32B
    int*   offs     = (int*)(ws + (512 << 10) + 64);  // 32B
    int*   srcIdx   = (int*)(ws + (512 << 10) + 128); // 128K

    moe_router <<<NCHUNK, 256, 0, stream>>>(x, Wr, br, best, bestp, chunkCnt);
    moe_scan   <<<NEXP,   256, 0, stream>>>(chunkCnt, chunkPre, totals);
    moe_offsets<<<1,       32, 0, stream>>>(totals, offs);
    moe_scatter<<<NCHUNK,  32, 0, stream>>>(best, chunkPre, offs, srcIdx);
    moe_gemm   <<<dim3(T_TOK / BM, HDIM / BN), 256, 0, stream>>>(
        x, We, be, srcIdx, bestp, out);
}
